// CrossAttentionFusion_59931973648837
// MI455X (gfx1250) — matrix operand order repaired and checked by reading, not compiled, not run on hardware
//
#include <hip/hip_runtime.h>
#include <hip/hip_bf16.h>

// CrossAttentionFusion on gfx1250 (MI455X):
//  pass 0: fp32 -> bf16 conversion (feats elementwise; weights pre-swizzled
//          into WMMA B-fragment tiles: tile(kt,nt) -> lane-contiguous 32B rows)
//  pass 1: fused QKV projection (v_wmma_f32_16x16x32_bf16, async-to-LDS
//          double-buffered activation staging) + cross-head softmax attention
//          (packed bf16 VALU), context tile -> workspace (bf16)
//  pass 2: context @ Wo + bo -> fp32 output (WMMA, same async staging)

#define DIM   1024
#define BROWS 16384
#define NHEAD 8
#define HD    128
#define MT    32      // rows per workgroup tile
#define KC    64      // K chunk per stage step (2 WMMA-K per step)
#define NPW   128     // output columns per wave (= one head in pass 1)
#define BLOCK 256     // 8 waves (wave32)

typedef __bf16 bf16_t;
typedef __attribute__((ext_vector_type(4)))  __bf16 v4bf;
typedef __attribute__((ext_vector_type(8)))  __bf16 v8bf;
typedef __attribute__((ext_vector_type(16))) __bf16 v16bf;
typedef __attribute__((ext_vector_type(4)))  float  v4f;
typedef __attribute__((ext_vector_type(8)))  float  v8f;

__device__ __forceinline__ v16bf cat16(v8bf lo, v8bf hi) {
  return __builtin_shufflevector(lo, hi, 0,1,2,3,4,5,6,7,8,9,10,11,12,13,14,15);
}

// CDNA5 async direct-to-LDS copy (ASYNCcnt-tracked, no VGPR round trip).
// Flat LDS pointers keep the byte offset in the low 32 bits (aperture in
// [63:32]), so truncating the flat address gives the LDS address VGPR.
__device__ __forceinline__ void async_ld_b128(void* lds_ptr, const void* gptr) {
  unsigned int lds = (unsigned int)(unsigned long long)lds_ptr;
  asm volatile("global_load_async_to_lds_b128 %0, %1, off"
               :: "v"(lds), "v"(gptr) : "memory");
}
__device__ __forceinline__ void wait_async0() {
  asm volatile("s_wait_asynccnt 0x0" ::: "memory");
}

// ---------------------------------------------------------------------------
// fp32 -> bf16 elementwise (feats). Grid sized exactly: n % (BLOCK*4) == 0.
__global__ void __launch_bounds__(BLOCK)
cvt_bf16_kernel(const float* __restrict__ src, bf16_t* __restrict__ dst) {
  const size_t i = ((size_t)blockIdx.x * BLOCK + threadIdx.x) * 4;
  v4f f = *(const v4f*)&src[i];
  v4bf o;
#pragma unroll
  for (int e = 0; e < 4; ++e) o[e] = (bf16_t)f[e];
  *(v4bf*)&dst[i] = o;
}

// ---------------------------------------------------------------------------
// Weight fp32 [k][n] row-major -> bf16 swizzled into WMMA B-fragment tiles.
// Tile (kt = k/32, nt = n/16) holds 32x16; within a tile, lane l owns column
// n = l%16 and 16 k values {0..7,16..23} + 8*(l/16) (32 bytes).
__global__ void __launch_bounds__(BLOCK)
wcvt_kernel(const float* __restrict__ src, bf16_t* __restrict__ dst) {
  const int t  = blockIdx.x * BLOCK + threadIdx.x;  // DIM*DIM/16 threads
  const int tile = t >> 5, ln = t & 31;
  const int k0 = (tile >> 6) * 32 + (ln >> 4) * 8;
  const int n  = (tile & 63) * 16 + (ln & 15);
  v16bf o;
#pragma unroll
  for (int s = 0; s < 16; ++s)
    o[s] = (bf16_t)src[(size_t)(k0 + (s & 7) + (s >> 3) * 16) * DIM + n];
  const size_t off = (size_t)tile * 512 + (size_t)ln * 16;
  *(v16bf*)&dst[off] = o;
}

// ---------------------------------------------------------------------------
// Issue the async stage of the [MT][KC] activation chunk starting at column kc
// into stage_buf (each of 256 threads copies 16 bytes).
__device__ __forceinline__ void stage_chunk(const bf16_t* __restrict__ src,
                                            int row0, int kc,
                                            bf16_t* __restrict__ stage_buf) {
  const int idx = threadIdx.x * 8;            // 256 thr * 8 elems = 2048
  const int r = idx >> 6, c = idx & 63;
  async_ld_b128(stage_buf + idx, &src[(size_t)(row0 + r) * DIM + kc + c]);
}

// ---------------------------------------------------------------------------
// One projection GEMM sweep: rows [row0, row0+MT) of src(bf16 [BROWS][DIM])
// times swizzled weight Wt, plus fp32 bias, result bf16 -> out_lds [MT][DIM].
// stage points to a double buffer of 2 * MT*KC bf16.
__device__ __forceinline__ void proj_gemm(
    const bf16_t* __restrict__ src, const bf16_t* __restrict__ Wt,
    const float* __restrict__ bias, bf16_t* __restrict__ out_lds,
    bf16_t* __restrict__ stage, int row0)
{
  const int lane = threadIdx.x & 31;
  const int wave = threadIdx.x >> 5;
  const int half = lane >> 4;   // 0: lanes 0-15, 1: lanes 16-31
  const int l16  = lane & 15;

  v8f acc[2][8];
#pragma unroll
  for (int mt = 0; mt < 2; ++mt)
#pragma unroll
    for (int nt = 0; nt < 8; ++nt)
      acc[mt][nt] = {};

  stage_chunk(src, row0, 0, stage);           // prefetch chunk 0 -> buf 0

  for (int kc = 0; kc < DIM; kc += KC) {
    bf16_t* cur = stage + ((kc >> 6) & 1) * (MT * KC);
    wait_async0();          // own async stores to LDS complete
    __syncthreads();        // everyone's stores complete; prev reads done
    if (kc + KC < DIM)      // prefetch next chunk into the other buffer
      stage_chunk(src, row0, kc + KC,
                  stage + (((kc >> 6) + 1) & 1) * (MT * KC));

#pragma unroll
    for (int ks = 0; ks < 2; ++ks) {          // two WMMA-K chunks per stage
      // A fragments (ISA 16-bit A layout: lane=M%16, K offsets {0,16}+8*half)
      v16bf a[2];
#pragma unroll
      for (int mt = 0; mt < 2; ++mt) {
        const bf16_t* ap = cur + (mt * 16 + l16) * KC + ks * 32 + half * 8;
        a[mt] = cat16(*(const v8bf*)ap, *(const v8bf*)(ap + 16));
      }
      // B fragments: pre-swizzled -> contiguous 32B per lane, coalesced
      const bf16_t* wbase =
          Wt + ((size_t)((kc >> 5) + ks) * 64 + (size_t)wave * 8) * 512 +
          (size_t)lane * 16;
#pragma unroll
      for (int nt = 0; nt < 8; ++nt) {
        v16bf b = *(const v16bf*)(wbase + nt * 512);
        acc[0][nt] = __builtin_amdgcn_wmma_f32_16x16x32_bf16(
            false, a[0], false, b, (short)0, acc[0][nt], false, false);
        acc[1][nt] = __builtin_amdgcn_wmma_f32_16x16x32_bf16(
            false, a[1], false, b, (short)0, acc[1][nt], false, false);
      }
    }
  }

  // epilogue: bias add, fp32 -> bf16, scatter into LDS [MT][DIM]
#pragma unroll
  for (int nt = 0; nt < 8; ++nt) {
    const int n  = wave * NPW + nt * 16 + l16;
    const float bs = bias[n];
#pragma unroll
    for (int mt = 0; mt < 2; ++mt)
#pragma unroll
      for (int v = 0; v < 8; ++v) {
        const int m = mt * 16 + v + half * 8;   // C-layout: M = v + 8*half
        out_lds[m * DIM + n] = (bf16_t)(acc[mt][nt][v] + bs);
      }
  }
}

// ---------------------------------------------------------------------------
// Pass 1: fused Q/K/V projection + cross-head attention for a 32-row tile.
__global__ void __launch_bounds__(BLOCK)
qkv_attn_kernel(const bf16_t* __restrict__ f1, const bf16_t* __restrict__ f2,
                const bf16_t* __restrict__ wq, const bf16_t* __restrict__ wk,
                const bf16_t* __restrict__ wv,
                const float* __restrict__ bq, const float* __restrict__ bk,
                const float* __restrict__ bv,
                bf16_t* __restrict__ ctx)
{
  extern __shared__ __align__(16) char smem[];
  bf16_t* q_lds = (bf16_t*)smem;             // [MT][DIM]
  bf16_t* k_lds = q_lds + MT * DIM;          // [MT][DIM]
  bf16_t* v_lds = k_lds + MT * DIM;          // [MT][DIM]
  bf16_t* stage = v_lds + MT * DIM;          // [2][MT][KC]

  const int row0 = blockIdx.x * MT;

  proj_gemm(f2, wk, bk, k_lds, stage, row0);
  __syncthreads();
  proj_gemm(f2, wv, bv, v_lds, stage, row0);
  __syncthreads();
  proj_gemm(f1, wq, bq, q_lds, stage, row0);
  __syncthreads();

  // Cross-head attention: thread = (sample s, query head h). 32*8 = 256.
  const int tid = threadIdx.x;
  const int s = tid >> 3;
  const int h = tid & 7;

  v8bf qv[16];
#pragma unroll
  for (int i = 0; i < 16; ++i)
    qv[i] = *(const v8bf*)&q_lds[s * DIM + h * HD + i * 8];

  // scores: packed bf16 FMA (v_pk_fma_bf16 path), f32 horizontal reduce
  float sc[8];
#pragma unroll
  for (int g = 0; g < 8; ++g) {
    const bf16_t* kr = &k_lds[s * DIM + g * HD];
    v8bf sa = {};
#pragma unroll
    for (int i = 0; i < 16; ++i)
      sa = qv[i] * (*(const v8bf*)&kr[i * 8]) + sa;
    float d = 0.f;
#pragma unroll
    for (int e = 0; e < 8; ++e) d += (float)sa[e];
    sc[g] = d * 0.08838834764831845f;  // 1/sqrt(HD)
  }
  float mx = sc[0];
#pragma unroll
  for (int g = 1; g < 8; ++g) mx = fmaxf(mx, sc[g]);
  float sum = 0.f;
#pragma unroll
  for (int g = 0; g < 8; ++g) { sc[g] = __expf(sc[g] - mx); sum += sc[g]; }
  const float inv = 1.f / sum;

  // attention weights as splatted bf16 vectors
  v8bf avv[8];
#pragma unroll
  for (int g = 0; g < 8; ++g) {
    const bf16_t ag = (bf16_t)(sc[g] * inv);
#pragma unroll
    for (int e = 0; e < 8; ++e) avv[g][e] = ag;
  }

  // context[s, h*128+d] = sum_g attn[g] * V[s, g*128+d]; overwrite Q slot
#pragma unroll
  for (int i = 0; i < 16; ++i) {
    v8bf cb = {};
#pragma unroll
    for (int g = 0; g < 8; ++g)
      cb = avv[g] * (*(const v8bf*)&v_lds[s * DIM + g * HD + i * 8]) + cb;
    *(v8bf*)&q_lds[s * DIM + h * HD + i * 8] = cb;
  }
  __syncthreads();

  // dump context tile to workspace (bf16)
#pragma unroll
  for (int it = 0; it < (MT * DIM) / (BLOCK * 8); ++it) {
    const int idx = (it * BLOCK + tid) * 8;
    *(v8bf*)&ctx[(size_t)row0 * DIM + idx] = *(const v8bf*)&q_lds[idx];
  }
}

// ---------------------------------------------------------------------------
// Pass 2: out = context @ Wo + bo (fp32 output)
__global__ void __launch_bounds__(BLOCK)
out_proj_kernel(const bf16_t* __restrict__ ctx, const bf16_t* __restrict__ wo,
                const float* __restrict__ bo, float* __restrict__ out)
{
  __shared__ __align__(16) bf16_t stage[2 * MT * KC];
  const int lane = threadIdx.x & 31;
  const int wave = threadIdx.x >> 5;
  const int half = lane >> 4;
  const int l16  = lane & 15;
  const int row0 = blockIdx.x * MT;

  v8f acc[2][8];
#pragma unroll
  for (int mt = 0; mt < 2; ++mt)
#pragma unroll
    for (int nt = 0; nt < 8; ++nt)
      acc[mt][nt] = {};

  stage_chunk(ctx, row0, 0, stage);

  for (int kc = 0; kc < DIM; kc += KC) {
    bf16_t* cur = stage + ((kc >> 6) & 1) * (MT * KC);
    wait_async0();
    __syncthreads();
    if (kc + KC < DIM)
      stage_chunk(ctx, row0, kc + KC,
                  stage + (((kc >> 6) + 1) & 1) * (MT * KC));

#pragma unroll
    for (int ks = 0; ks < 2; ++ks) {
      v16bf a[2];
#pragma unroll
      for (int mt = 0; mt < 2; ++mt) {
        const bf16_t* ap = cur + (mt * 16 + l16) * KC + ks * 32 + half * 8;
        a[mt] = cat16(*(const v8bf*)ap, *(const v8bf*)(ap + 16));
      }
      const bf16_t* wbase =
          wo + ((size_t)((kc >> 5) + ks) * 64 + (size_t)wave * 8) * 512 +
          (size_t)lane * 16;
#pragma unroll
      for (int nt = 0; nt < 8; ++nt) {
        v16bf b = *(const v16bf*)(wbase + nt * 512);
        acc[0][nt] = __builtin_amdgcn_wmma_f32_16x16x32_bf16(
            false, a[0], false, b, (short)0, acc[0][nt], false, false);
        acc[1][nt] = __builtin_amdgcn_wmma_f32_16x16x32_bf16(
            false, a[1], false, b, (short)0, acc[1][nt], false, false);
      }
    }
  }

#pragma unroll
  for (int nt = 0; nt < 8; ++nt) {
    const int n  = wave * NPW + nt * 16 + l16;
    const float bs = bo[n];
#pragma unroll
    for (int mt = 0; mt < 2; ++mt)
#pragma unroll
      for (int v = 0; v < 8; ++v) {
        const int m = mt * 16 + v + half * 8;
        out[(size_t)(row0 + m) * DIM + n] = acc[mt][nt][v] + bs;
      }
  }
}

// ---------------------------------------------------------------------------
extern "C" void kernel_launch(void* const* d_in, const int* in_sizes, int n_in,
                              void* d_out, int out_size, void* d_ws, size_t ws_size,
                              hipStream_t stream) {
  (void)in_sizes; (void)n_in; (void)out_size; (void)ws_size;
  const float* feat1 = (const float*)d_in[0];
  const float* feat2 = (const float*)d_in[1];
  const float* Wq = (const float*)d_in[2];
  const float* bq = (const float*)d_in[3];
  const float* Wk = (const float*)d_in[4];
  const float* bk = (const float*)d_in[5];
  const float* Wv = (const float*)d_in[6];
  const float* bv = (const float*)d_in[7];
  const float* Wo = (const float*)d_in[8];
  const float* bo = (const float*)d_in[9];
  float* out = (float*)d_out;

  const size_t F = (size_t)BROWS * DIM;
  const size_t W = (size_t)DIM * DIM;
  bf16_t* ws  = (bf16_t*)d_ws;
  bf16_t* f1b = ws;
  bf16_t* f2b = ws + F;
  bf16_t* wqb = ws + 2 * F;
  bf16_t* wkb = wqb + W;
  bf16_t* wvb = wkb + W;
  bf16_t* wob = wvb + W;
  bf16_t* ctx = wob + W;

  // pass 0: conversions
  cvt_bf16_kernel<<<(int)(F / (BLOCK * 4)), BLOCK, 0, stream>>>(feat1, f1b);
  cvt_bf16_kernel<<<(int)(F / (BLOCK * 4)), BLOCK, 0, stream>>>(feat2, f2b);
  wcvt_kernel<<<(int)(W / 16 / BLOCK), BLOCK, 0, stream>>>(Wq, wqb);
  wcvt_kernel<<<(int)(W / 16 / BLOCK), BLOCK, 0, stream>>>(Wk, wkb);
  wcvt_kernel<<<(int)(W / 16 / BLOCK), BLOCK, 0, stream>>>(Wv, wvb);
  wcvt_kernel<<<(int)(W / 16 / BLOCK), BLOCK, 0, stream>>>(Wo, wob);

  // pass 1: fused QKV + cross-head attention
  const int smem1 = (3 * MT * DIM + 2 * MT * KC) * (int)sizeof(bf16_t); // 204800
  hipFuncSetAttribute(reinterpret_cast<const void*>(qkv_attn_kernel),
                      hipFuncAttributeMaxDynamicSharedMemorySize, smem1);
  qkv_attn_kernel<<<BROWS / MT, BLOCK, smem1, stream>>>(
      f1b, f2b, wqb, wkb, wvb, bq, bk, bv, ctx);

  // pass 2: output projection
  out_proj_kernel<<<BROWS / MT, BLOCK, 0, stream>>>(ctx, wob, bo, out);
}
